// RelativeGlobalAttention_68771016343576
// MI455X (gfx1250) — matrix operand order repaired and checked by reading, not compiled, not run on hardware
//
#include <hip/hip_runtime.h>

// ---------------------------------------------------------------------------
// Relative Global Attention (music-transformer style) for MI455X / gfx1250.
//
// Roofline: output attn_w is 2*8*2048*2048 f32 = 268MB which MUST be written
// (tuple output), so the op is HBM-bound (~268MB * ~4 touches / 23.3TB/s).
// All GEMMs are done with v_wmma_f32_16x16x32_bf16 (CDNA5's fast dense path);
// fp32 WMMA only exists as 16x16x4 and would be VALU-roofline limited.
//
// Skew identity (derived from the pad/reshape trick in the reference):
//   for t <= s:  Srel[s,t] = QE[s, L-1-s+t],  QE[s,l] = q_s . e_l + Eb[h,l]
// so we stage QE rows in LDS (32 x 2048 f32 = 256KB, fits the 320KB WGP LDS)
// and read them shifted — no materialized skew tensor.
// ---------------------------------------------------------------------------

typedef __attribute__((ext_vector_type(16))) __bf16 v16bf;
typedef __attribute__((ext_vector_type(8)))  __bf16 bf16x8;
typedef __attribute__((ext_vector_type(8)))  float  v8f;

#define DEV static __device__ __forceinline__

constexpr int   C_B  = 2;
constexpr int   C_H  = 8;
constexpr int   C_S  = 2048;
constexpr int   C_D  = 512;
constexpr int   C_DH = 64;
constexpr float C_SCALE = 0.125f;   // 1/sqrt(64)

// ---- WMMA fragment helpers (wave32 layouts per CDNA5 ISA 7.12.2) ----------

// A (16x32 bf16): lane&15 = row M; lanes<16 hold K {0..7,16..23}, lanes>=16
// hold K {8..15,24..31}. Row-major source, K contiguous -> two 16B loads.
DEV v16bf load_a16(const __bf16* base, int ld, int r0, int k0, int lane) {
  int r = lane & 15, hi = lane >> 4;
  const __bf16* p = base + (size_t)(r0 + r) * ld + k0 + hi * 8;
  bf16x8 c0 = *(const bf16x8*)p;
  bf16x8 c1 = *(const bf16x8*)(p + 16);
  v16bf a;
#pragma unroll
  for (int i = 0; i < 8; ++i) { a[i] = c0[i]; a[i + 8] = c1[i]; }
  return a;
}

// B (32x16 bf16) where B[k,n] = src[(n0+n)*ld + k0+k] (i.e. src holds B^T
// row-major, K contiguous). lane&15 = col N; lanes<16 K {0..7,16..23},
// lanes>=16 K {8..15,24..31} -> two 16B loads.
DEV v16bf load_bt16(const __bf16* base, int ld, int n0, int k0, int lane) {
  int n = lane & 15, hi = lane >> 4;
  const __bf16* p = base + (size_t)(n0 + n) * ld + k0 + hi * 8;
  bf16x8 c0 = *(const bf16x8*)p;
  bf16x8 c1 = *(const bf16x8*)(p + 16);
  v16bf b;
#pragma unroll
  for (int i = 0; i < 8; ++i) { b[i] = c0[i]; b[i + 8] = c1[i]; }
  return b;
}

DEV v8f wmma_bf(v16bf a, v16bf b, v8f c) {
  // (neg_a, A, neg_b, B, c_mod, C, reuse_a, reuse_b)
  return __builtin_amdgcn_wmma_f32_16x16x32_bf16(false, a, false, b,
                                                 (short)0, c, false, false);
}

// ---- f32 -> bf16 conversion ------------------------------------------------

__global__ __launch_bounds__(256) void cvt_bf16(const float* __restrict__ s,
                                                __bf16* __restrict__ d, int n) {
  int i = blockIdx.x * 256 + threadIdx.x;
  if (i < n) d[i] = (__bf16)s[i];
}

// ---- QKV projection: qkv = x @ Wqkv^T + bqkv, scattered to q/k/vT ---------
// M=4096 (B*S), N=1536, K=512. 8 waves/block, one 16x16 tile per wave.
// q,k stored (B,H,S,DH) bf16; v stored transposed (B,H,DH,S) bf16 so the
// P@V B-fragment load is contiguous along t.

__global__ __launch_bounds__(256) void qkv_gemm(
    const __bf16* __restrict__ xb, const __bf16* __restrict__ wb,
    const float* __restrict__ bias, __bf16* __restrict__ qb,
    __bf16* __restrict__ kb, __bf16* __restrict__ vT) {
  int lane = threadIdx.x & 31, wid = threadIdx.x >> 5;
  int gt = blockIdx.x * 8 + wid;            // 24576 tiles total
  int mt = gt / 96, nt = gt % 96;
  int m0 = mt * 16, n0 = nt * 16;
  v8f acc = {};
#pragma unroll 4
  for (int kc = 0; kc < C_D; kc += 32) {
    v16bf a = load_a16(xb, C_D, m0, kc, lane);
    v16bf b = load_bt16(wb, C_D, n0, kc, lane);
    acc = wmma_bf(a, b, acc);
  }
  int hi = lane >> 4, nn = lane & 15;
  int n = n0 + nn;
  float bv = bias[n];
  int which = n >> 9, h = (n >> 6) & 7, d = n & 63;
#pragma unroll
  for (int j = 0; j < 8; ++j) {
    int m = m0 + j + 8 * hi;
    int bb = m >> 11, s = m & 2047;
    __bf16 val = (__bf16)(acc[j] + bv);
    if (which == 2) {
      vT[((size_t)(bb * C_H + h) * C_DH + d) * C_S + s] = val;
    } else {
      size_t idx = ((size_t)(bb * C_H + h) * C_S + s) * C_DH + d;
      if (which == 0) qb[idx]; else kb[idx];
      (which == 0 ? qb : kb)[idx] = val;
    }
  }
}

// ---- attention: one workgroup = (b, h, 32-row tile); 8 waves --------------

__global__ __launch_bounds__(256) void attn_kernel(
    const __bf16* __restrict__ qb, const __bf16* __restrict__ kb,
    const __bf16* __restrict__ vT, const __bf16* __restrict__ eb,
    const float* __restrict__ Ebias, float* __restrict__ attnw_all,
    __bf16* __restrict__ attnB) {
  extern __shared__ char smem_raw[];
  float* qe     = (float*)smem_raw;       // [32][2048] f32 = 256KB
  float* stat   = qe + 32 * C_S;          // [8][32] per-wave row maxima
  float* rowmax = stat + 256;             // [32]
  float* rinv   = rowmax + 32;            // [32] 1/rowsum

  int lane = threadIdx.x & 31, wid = threadIdx.x >> 5;
  int hi = lane >> 4, nn = lane & 15;
  int bid = blockIdx.x;
  int rt = bid & 63, h = (bid >> 6) & 7, b = bid >> 9;
  int s0 = rt * 32;

  const __bf16* qp = qb + ((size_t)(b * C_H + h) * C_S + s0) * C_DH;
  const __bf16* kp = kb + (size_t)(b * C_H + h) * C_S * C_DH;
  const __bf16* vp = vT + (size_t)(b * C_H + h) * C_DH * C_S;
  const __bf16* ep = eb + (size_t)h * C_S * C_DH;
  const float*  ebias = Ebias + h * C_S;
  float* pw = attnw_all + ((size_t)(b * C_H + h) * C_S + s0) * (size_t)C_S;

  // q fragments for both 16-row subtiles, both K-halves (DH=64 -> 2 x K32)
  v16bf aq[2][2];
#pragma unroll
  for (int sub = 0; sub < 2; ++sub)
#pragma unroll
    for (int kc = 0; kc < 2; ++kc)
      aq[sub][kc] = load_a16(qp, C_DH, sub * 16, kc * 32, lane);

  // ---- Stage QE: only the l-band [2016-s0, 2048) is ever read (skew) ----
  int lqt0 = 126 - 2 * rt;
  for (int ct = lqt0 + wid; ct < 128; ct += 8) {
    int l0 = ct * 16;
    v16bf b0 = load_bt16(ep, C_DH, l0, 0, lane);
    v16bf b1 = load_bt16(ep, C_DH, l0, 32, lane);
    float ea = ebias[l0 + nn];
#pragma unroll
    for (int sub = 0; sub < 2; ++sub) {
      v8f acc = {};
      acc = wmma_bf(aq[sub][0], b0, acc);
      acc = wmma_bf(aq[sub][1], b1, acc);
#pragma unroll
      for (int j = 0; j < 8; ++j) {
        int r = sub * 16 + j + 8 * hi;
        qe[r * C_S + l0 + nn] = acc[j] + ea;
      }
    }
  }
  __syncthreads();

  // ---- P1: logits = (q.k^T + skewed QE) * scale, causal mask; row max ----
  float lmax[16];
#pragma unroll
  for (int i = 0; i < 16; ++i) lmax[i] = -3.4e38f;
  int tlim = 2 * rt + 1;                       // last col-tile with t <= s_max
  for (int ct = wid; ct <= tlim; ct += 8) {
    int t0 = ct * 16;
    __builtin_prefetch(kp + (size_t)(t0 + 128) * C_DH, 0, 1);
    v16bf b0 = load_bt16(kp, C_DH, t0, 0, lane);
    v16bf b1 = load_bt16(kp, C_DH, t0, 32, lane);
    int t = t0 + nn;
#pragma unroll
    for (int sub = 0; sub < 2; ++sub) {
      v8f acc = {};
      acc = wmma_bf(aq[sub][0], b0, acc);
      acc = wmma_bf(aq[sub][1], b1, acc);
#pragma unroll
      for (int j = 0; j < 8; ++j) {
        int r = sub * 16 + j + 8 * hi;
        int s = s0 + r;
        float val = (t <= s)
            ? (acc[j] + qe[r * C_S + (2047 - s + t)]) * C_SCALE
            : -1.0e9f;
        lmax[sub * 8 + j] = fmaxf(lmax[sub * 8 + j], val);
        pw[(size_t)r * C_S + t] = val;
      }
    }
  }
  // reduce maxima across the 16 lanes holding the same rows, then cross-wave
#pragma unroll
  for (int sub = 0; sub < 2; ++sub)
#pragma unroll
    for (int j = 0; j < 8; ++j) {
      float v = lmax[sub * 8 + j];
      for (int m = 1; m < 16; m <<= 1) v = fmaxf(v, __shfl_xor(v, m, 16));
      if (nn == 0) stat[wid * 32 + sub * 16 + j + 8 * hi] = v;
    }
  __threadfence_block();
  __syncthreads();
  if (threadIdx.x < 32) {
    float m = -3.4e38f;
#pragma unroll
    for (int w = 0; w < 8; ++w) m = fmaxf(m, stat[w * 32 + threadIdx.x]);
    rowmax[threadIdx.x] = m;
  }
  __syncthreads();

  // ---- P2: p = exp(x - max) over the FULL 32x2048 block (masked -> 0) ----
  {
    int r = threadIdx.x >> 3;                 // 8 threads per row
    int seg = (threadIdx.x & 7) * 256;
    int s = s0 + r;
    float m = rowmax[r];
    float sum = 0.f;
    float* prow = pw + (size_t)r * C_S;
    for (int c = seg; c < seg + 256; c += 4) {
      float4 x = *(const float4*)(prow + c);
      float4 p;
      p.x = (c + 0 <= s) ? __expf(x.x - m) : 0.f;
      p.y = (c + 1 <= s) ? __expf(x.y - m) : 0.f;
      p.z = (c + 2 <= s) ? __expf(x.z - m) : 0.f;
      p.w = (c + 3 <= s) ? __expf(x.w - m) : 0.f;
      sum += p.x + p.y + p.z + p.w;
      *(float4*)(prow + c) = p;
    }
    for (int m2 = 1; m2 < 8; m2 <<= 1) sum += __shfl_xor(sum, m2, 8);
    if ((threadIdx.x & 7) == 0) rinv[r] = 1.0f / sum;
  }
  __threadfence_block();
  __syncthreads();

  // ---- P3: attn = (P @ V) * (1/sum); wave = (sub, d-tile) ----------------
  {
    int sub = wid >> 2, dt = wid & 3;
    int d0 = dt * 16;
    v8f acc = {};
    int r = sub * 16 + nn;                    // A row (local)
    const float* prow = pw + (size_t)r * C_S;
    for (int kt = 0; kt <= rt; ++kt) {        // K covers all t <= s_max
      int kb0 = kt * 32;
      const float* pa = prow + kb0 + hi * 8;
      float4 f0 = *(const float4*)pa;
      float4 f1 = *(const float4*)(pa + 4);
      float4 f2 = *(const float4*)(pa + 16);
      float4 f3 = *(const float4*)(pa + 20);
      v16bf a;
      a[0]  = (__bf16)f0.x; a[1]  = (__bf16)f0.y; a[2]  = (__bf16)f0.z; a[3]  = (__bf16)f0.w;
      a[4]  = (__bf16)f1.x; a[5]  = (__bf16)f1.y; a[6]  = (__bf16)f1.z; a[7]  = (__bf16)f1.w;
      a[8]  = (__bf16)f2.x; a[9]  = (__bf16)f2.y; a[10] = (__bf16)f2.z; a[11] = (__bf16)f2.w;
      a[12] = (__bf16)f3.x; a[13] = (__bf16)f3.y; a[14] = (__bf16)f3.z; a[15] = (__bf16)f3.w;
      v16bf bf = load_bt16(vp, C_S, d0, kb0, lane);
      acc = wmma_bf(a, bf, acc);
    }
#pragma unroll
    for (int j = 0; j < 8; ++j) {
      int rl = sub * 16 + j + 8 * hi;
      int s = s0 + rl;
      float v = acc[j] * rinv[rl];
      attnB[((size_t)(b * C_S + s) * C_H + h) * C_DH + d0 + nn] = (__bf16)v;
    }
  }
  __syncthreads();

  // ---- P4: normalize attn_w in place (0 region: 0*inv = 0, harmless) ----
  {
    int r = threadIdx.x >> 3;
    int seg = (threadIdx.x & 7) * 256;
    float iv = rinv[r];
    float* prow = pw + (size_t)r * C_S;
    for (int c = seg; c < seg + 256; c += 4) {
      float4 x = *(const float4*)(prow + c);
      x.x *= iv; x.y *= iv; x.z *= iv; x.w *= iv;
      *(float4*)(prow + c) = x;
    }
  }
}

// ---- output projection: out = attn(BSxD) @ Wfc^T + bfc --------------------

__global__ __launch_bounds__(256) void outproj(
    const __bf16* __restrict__ ab, const __bf16* __restrict__ wb,
    const float* __restrict__ bias, float* __restrict__ out) {
  int lane = threadIdx.x & 31, wid = threadIdx.x >> 5;
  int gt = blockIdx.x * 8 + wid;            // 8192 tiles
  int mt = gt >> 5, nt = gt & 31;
  int m0 = mt * 16, n0 = nt * 16;
  v8f acc = {};
#pragma unroll 4
  for (int kc = 0; kc < C_D; kc += 32) {
    v16bf a = load_a16(ab, C_D, m0, kc, lane);
    v16bf b = load_bt16(wb, C_D, n0, kc, lane);
    acc = wmma_bf(a, b, acc);
  }
  int hi = lane >> 4, nn = lane & 15;
  int n = n0 + nn;
  float bv = bias[n];
#pragma unroll
  for (int j = 0; j < 8; ++j) {
    int m = m0 + j + 8 * hi;
    out[(size_t)m * C_D + n] = acc[j] + bv;
  }
}

// ---------------------------------------------------------------------------

extern "C" void kernel_launch(void* const* d_in, const int* in_sizes, int n_in,
                              void* d_out, int out_size, void* d_ws,
                              size_t ws_size, hipStream_t stream) {
  const float* x    = (const float*)d_in[0];
  // d_in[1] = causal mask (triu k=1) — structure is known, not needed.
  const float* Wqkv = (const float*)d_in[2];
  const float* bqkv = (const float*)d_in[3];
  const float* Wfc  = (const float*)d_in[4];
  const float* bfc  = (const float*)d_in[5];
  const float* E    = (const float*)d_in[6];
  const float* Eb   = (const float*)d_in[7];

  char* ws = (char*)d_ws;
  size_t o = 0;
  __bf16* xb  = (__bf16*)(ws + o); o += (size_t)C_B * C_S * C_D * 2;       // 4MB
  __bf16* wqb = (__bf16*)(ws + o); o += (size_t)3 * C_D * C_D * 2;        // 1.5MB
  __bf16* wfb = (__bf16*)(ws + o); o += (size_t)C_D * C_D * 2;            // 0.5MB
  __bf16* ebb = (__bf16*)(ws + o); o += (size_t)C_H * C_S * C_DH * 2;     // 2MB
  __bf16* qb  = (__bf16*)(ws + o); o += (size_t)C_B * C_H * C_S * C_DH * 2;
  __bf16* kb  = (__bf16*)(ws + o); o += (size_t)C_B * C_H * C_S * C_DH * 2;
  __bf16* vT  = (__bf16*)(ws + o); o += (size_t)C_B * C_H * C_S * C_DH * 2;
  __bf16* aB  = (__bf16*)(ws + o); o += (size_t)C_B * C_S * C_D * 2;      // ~25MB total

  float* out   = (float*)d_out;
  float* attnw = out + (size_t)C_B * C_S * C_D;

  int n_x = C_B * C_S * C_D, n_wq = 3 * C_D * C_D, n_wf = C_D * C_D,
      n_e = C_H * C_S * C_DH;
  cvt_bf16<<<(n_x + 255) / 256, 256, 0, stream>>>(x, xb, n_x);
  cvt_bf16<<<(n_wq + 255) / 256, 256, 0, stream>>>(Wqkv, wqb, n_wq);
  cvt_bf16<<<(n_wf + 255) / 256, 256, 0, stream>>>(Wfc, wfb, n_wf);
  cvt_bf16<<<(n_e + 255) / 256, 256, 0, stream>>>(E, ebb, n_e);

  qkv_gemm<<<3072, 256, 0, stream>>>(xb, wqb, bqkv, qb, kb, vT);

  size_t smem = (size_t)32 * C_S * 4 + (256 + 32 + 32) * 4;  // 257.25 KB
  attn_kernel<<<C_B * C_H * (C_S / 32), 256, smem, stream>>>(
      qb, kb, vT, ebb, Eb, attnw, aB);

  outproj<<<1024, 256, 0, stream>>>(aB, wfb, bfc, out);
}
